// FusionAttentionGNN_Encoder_38379827757430
// MI455X (gfx1250) — matrix operand order repaired and checked by reading, compiled, not run on hardware
//
#include <hip/hip_runtime.h>
#include <hip/hip_bf16.h>
#include <stdint.h>

// Problem constants (fixed by the reference).
#define U_CNT 8192
#define I_CNT 8192
#define N_CNT 16384
#define D_CNT 128
#define E_CNT 524288
#define K_TOP 32
#define ND (N_CNT * D_CNT)

typedef __attribute__((ext_vector_type(16))) __bf16 v16bf;
typedef __attribute__((ext_vector_type(8)))  __bf16 v8bf;
typedef __attribute__((ext_vector_type(8)))  float  v8f;
typedef __attribute__((ext_vector_type(2)))  float  v2f;

// ---------------------------------------------------------------- elementwise
__global__ __launch_bounds__(256) void concat_kernel(
    const float* __restrict__ user, const float* __restrict__ item,
    float* __restrict__ x0) {
  int i = blockIdx.x * 256 + threadIdx.x;
  x0[i] = (i < U_CNT * D_CNT) ? user[i] : item[i - U_CNT * D_CNT];
}

__global__ __launch_bounds__(256) void zero_kernel(float* __restrict__ p) {
  p[blockIdx.x * 256 + threadIdx.x] = 0.f;
}

__global__ __launch_bounds__(256) void accum_kernel(
    float* __restrict__ dst, const float* __restrict__ src) {
  int i = blockIdx.x * 256 + threadIdx.x;
  dst[i] += src[i];
}

__global__ __launch_bounds__(256) void mean3_kernel(
    float* __restrict__ dst, const float* __restrict__ sum) {
  int i = blockIdx.x * 256 + threadIdx.x;
  dst[i] = sum[i] * (1.0f / 3.0f);
}

__global__ __launch_bounds__(256) void cvt_bf16_kernel(
    __bf16* __restrict__ dst, const float* __restrict__ src) {
  int i = blockIdx.x * 256 + threadIdx.x;
  dst[i] = (__bf16)src[i];
}

// ------------------------------------------------------------------- SpMM
// Edge-parallel scatter: one wave per edge, lane l handles dims [4l, 4l+4).
// Coalesced 512B row reads; hardware f32 atomic adds into the output.
__global__ __launch_bounds__(256) void spmm_kernel(
    const int* __restrict__ rows, const int* __restrict__ cols,
    const float* __restrict__ vals, const float* __restrict__ x,
    float* __restrict__ y) {
  int e = blockIdx.x * 8 + (threadIdx.x >> 5);
  int lane = threadIdx.x & 31;
  if (e >= E_CNT) return;
  int r = rows[e], c = cols[e];
  float v = vals[e];
  const float4 xv = *(const float4*)(x + (size_t)c * D_CNT + lane * 4);
  float* yb = y + (size_t)r * D_CNT + lane * 4;
  __hip_atomic_fetch_add(yb + 0, v * xv.x, __ATOMIC_RELAXED, __HIP_MEMORY_SCOPE_AGENT);
  __hip_atomic_fetch_add(yb + 1, v * xv.y, __ATOMIC_RELAXED, __HIP_MEMORY_SCOPE_AGENT);
  __hip_atomic_fetch_add(yb + 2, v * xv.z, __ATOMIC_RELAXED, __HIP_MEMORY_SCOPE_AGENT);
  __hip_atomic_fetch_add(yb + 3, v * xv.w, __ATOMIC_RELAXED, __HIP_MEMORY_SCOPE_AGENT);
}

// -------------------------------------------------- fused sim(ego@ego^T)+topk
// One block per 16 rows. 8 waves compute 16x16 score tiles via
// v_wmma_f32_16x16x32_bf16 into a 16x512 LDS chunk; per-row top-32 lists are
// maintained with a shuffle-reduced max pre-filter + stable serial insert.
#define SIM_CHUNK 512

__global__ __launch_bounds__(256) void simtopk_kernel(
    const __bf16* __restrict__ ego16, int* __restrict__ topk_out) {
  __shared__ float sc[16 * SIM_CHUNK];   // 32 KB score chunk
  __shared__ float topv[16][K_TOP];
  __shared__ int   topi[16][K_TOP];
  __shared__ int   rowflag[16];

  const int tid  = threadIdx.x;
  const int lane = tid & 31;
  const int wid  = tid >> 5;                 // 0..7
  const int row_base = blockIdx.x * 16;

  for (int i = tid; i < 16 * K_TOP; i += 256) {
    topv[i >> 5][i & 31] = -3.4e38f;
    topi[i >> 5][i & 31] = 0;
  }

  // A fragments: 16x32 bf16, lane = row (lane&15), K-half by lane>=16.
  const int am    = lane & 15;
  const int ahalf = (lane >> 4) * 16;
  v16bf afrag[4];
  const __bf16* arow = ego16 + (size_t)(row_base + am) * D_CNT;
#pragma unroll
  for (int j = 0; j < 4; ++j) {
    const __bf16* ap = arow + j * 32 + ahalf;
    v8bf lo = *(const v8bf*)(ap);
    v8bf hi = *(const v8bf*)(ap + 8);
    v16bf a;
#pragma unroll
    for (int t = 0; t < 8; ++t) { a[t] = lo[t]; a[t + 8] = hi[t]; }
    afrag[j] = a;
  }
  __syncthreads();

  const int bn    = lane & 15;               // column within tile
  const int bkoff = (lane >> 4) * 16;        // K-half for B

  for (int chunk = 0; chunk < N_CNT / SIM_CHUNK; ++chunk) {
    const int col_chunk = chunk * SIM_CHUNK;
    if (chunk + 1 < N_CNT / SIM_CHUNK)
      __builtin_prefetch(ego16 + (size_t)(col_chunk + SIM_CHUNK) * D_CNT, 0, 1);

#pragma unroll
    for (int t = 0; t < 4; ++t) {
      const int tile = wid * 4 + t;          // 32 tiles per chunk, 4 per wave
      const int col_base = col_chunk + tile * 16;
      const __bf16* brow = ego16 + (size_t)(col_base + bn) * D_CNT + bkoff;
      // Preload all four K-step B fragments so the loads clause together
      // and the MMA chain is not stalled per step.
      v16bf bfrag[4];
#pragma unroll
      for (int j = 0; j < 4; ++j) bfrag[j] = *(const v16bf*)(brow + j * 32);
      v8f acc = {0.f, 0.f, 0.f, 0.f, 0.f, 0.f, 0.f, 0.f};
#pragma unroll
      for (int j = 0; j < 4; ++j) {          // K = 128 = 4 x 32
        acc = __builtin_amdgcn_wmma_f32_16x16x32_bf16(
            false, afrag[j], false, bfrag[j], (short)0, acc, false, false);
      }
      const int roff = (lane >> 4) * 8;      // C layout: rows v+8*(lane>=16)
#pragma unroll
      for (int v = 0; v < 8; ++v)
        sc[(v + roff) * SIM_CHUNK + tile * 16 + bn] = acc[v];
    }
    __syncthreads();

    // Pre-filter: chunk row max vs current 32nd value.
    const int r   = tid >> 4;
    const int lin = tid & 15;
    float lm = -3.4e38f;
    for (int i = lin; i < SIM_CHUNK; i += 16)
      lm = fmaxf(lm, sc[r * SIM_CHUNK + i]);
#pragma unroll
    for (int m = 8; m; m >>= 1) lm = fmaxf(lm, __shfl_xor(lm, m, 16));
    if (lin == 0) rowflag[r] = (lm > topv[r][K_TOP - 1]) ? 1 : 0;
    __syncthreads();

    if (tid < 16 && rowflag[tid]) {
      const int rr = tid;
      float mn = topv[rr][K_TOP - 1];
      for (int i = 0; i < SIM_CHUNK; ++i) {
        float v = sc[rr * SIM_CHUNK + i];
        if (v > mn) {                        // strict > keeps JAX tie order
          int j = K_TOP - 1;
          while (j > 0 && topv[rr][j - 1] < v) {
            topv[rr][j] = topv[rr][j - 1];
            topi[rr][j] = topi[rr][j - 1];
            --j;
          }
          topv[rr][j] = v;
          topi[rr][j] = col_chunk + i;
          mn = topv[rr][K_TOP - 1];
        }
      }
    }
    __syncthreads();
  }

  for (int i = tid; i < 16 * K_TOP; i += 256)
    topk_out[(size_t)(row_base + (i >> 5)) * K_TOP + (i & 31)] = topi[i >> 5][i & 31];
}

// ------------------------------------------------- small fp32 GEMM (D=128)
// Y[N,128] = X @ op(W) (+bias) (+scale*extra) (tanh?) using exact-fp32
// v_wmma_f32_16x16x4_f32. 8 waves/block cover a 64x128 output tile.
__global__ __launch_bounds__(256) void gemm128_kernel(
    const float* __restrict__ X, const float* __restrict__ W,
    const float* __restrict__ bias, const float* __restrict__ extra,
    float extraScale, int transB, int doTanh, float* __restrict__ Y) {
  const int tid  = threadIdx.x;
  const int lane = tid & 31;
  const int wid  = tid >> 5;
  const int rt   = wid >> 1;                 // row tile 0..3
  const int ch   = wid & 1;                  // column half
  const int row_base = blockIdx.x * 64 + rt * 16;
  const int m    = lane & 15;
  const int koff = (lane >> 4) * 2;          // K-half for A/B
  const float* xrow = X + (size_t)(row_base + m) * D_CNT;

  for (int ct = 0; ct < 4; ++ct) {
    const int col_base = ch * 64 + ct * 16;
    const int j = col_base + m;
    v8f acc = {0.f, 0.f, 0.f, 0.f, 0.f, 0.f, 0.f, 0.f};
    for (int kb = 0; kb < D_CNT; kb += 4) {
      v2f a, b;
      a.x = xrow[kb + koff];
      a.y = xrow[kb + koff + 1];
      if (transB) {                          // B[k][j] = W[j][k]
        b.x = W[(size_t)j * D_CNT + kb + koff];
        b.y = W[(size_t)j * D_CNT + kb + koff + 1];
      } else {                               // B[k][j] = W[k][j]
        b.x = W[(size_t)(kb + koff) * D_CNT + j];
        b.y = W[(size_t)(kb + koff + 1) * D_CNT + j];
      }
      acc = __builtin_amdgcn_wmma_f32_16x16x4_f32(
          false, a, false, b, (short)0, acc, false, false);
    }
    const int roff = (lane >> 4) * 8;
#pragma unroll
    for (int v = 0; v < 8; ++v) {
      const int rr = row_base + roff + v;
      float val = acc[v];
      if (bias)  val += bias[j];
      if (extra) val += extraScale * extra[(size_t)rr * D_CNT + j];
      if (doTanh) val = tanhf(val);
      Y[(size_t)rr * D_CNT + j] = val;
    }
  }
}

// ------------------------------------------- gathered attention contraction
// One block (128 threads) per node. The 32 sample rows (16 KB) are gathered
// into LDS with GLOBAL_LOAD_ASYNC_TO_LDS_B128 (ASYNCcnt path, no VGPR
// round-trip), then scores[k] = p[n]·sample_k / sqrt(D) (q·bk is constant in
// k -> cancels in softmax) and c[n] = sum_k attn[k]*sample_k.
__global__ __launch_bounds__(128) void attn_kernel(
    const float* __restrict__ p, const float* __restrict__ ego,
    const int* __restrict__ topk, float* __restrict__ c) {
  __shared__ float smp[K_TOP * D_CNT];       // 16 KB gathered samples
  __shared__ float sc[K_TOP];
  __shared__ float wred[4];
  __shared__ int   sidx[K_TOP];
  const int n = blockIdx.x;
  const int d = threadIdx.x;
  if (d < K_TOP) sidx[d] = topk[(size_t)n * K_TOP + d];
  __syncthreads();

  // Async gather: 8 x b128 per thread covers all K_TOP*D floats.
  const uint32_t smp_lds = (uint32_t)(uintptr_t)&smp[0];
#pragma unroll
  for (int i = 0; i < 8; ++i) {
    const int flat = i * 512 + d * 4;        // float index into smp
    const int k    = flat >> 7;
    const int dim  = flat & (D_CNT - 1);
    const float* src = ego + (size_t)sidx[k] * D_CNT + dim;
    const uint32_t dst = smp_lds + (uint32_t)flat * 4u;
    asm volatile("global_load_async_to_lds_b128 %0, %1, off"
                 :: "v"(dst), "v"(src) : "memory");
  }
  asm volatile("s_wait_asynccnt 0x0" ::: "memory");
  __syncthreads();

  const float pd = p[(size_t)n * D_CNT + d];
  for (int k = 0; k < K_TOP; ++k) {
    float part = pd * smp[k * D_CNT + d];
#pragma unroll
    for (int off = 16; off; off >>= 1) part += __shfl_xor(part, off, 32);
    if ((d & 31) == 0) wred[d >> 5] = part;
    __syncthreads();
    if (d == 0)
      sc[k] = (wred[0] + wred[1] + wred[2] + wred[3]) * 0.08838834764831845f;
    __syncthreads();
  }
  if (d < K_TOP) {                           // softmax over 32 lanes (wave 0)
    float v = sc[d];
    float mx = v;
#pragma unroll
    for (int off = 16; off; off >>= 1) mx = fmaxf(mx, __shfl_xor(mx, off, 32));
    float e = expf(v - mx);
    float sm = e;
#pragma unroll
    for (int off = 16; off; off >>= 1) sm += __shfl_xor(sm, off, 32);
    sc[d] = e / sm;
  }
  __syncthreads();
  float acc = 0.f;
  for (int k = 0; k < K_TOP; ++k) acc += sc[k] * smp[k * D_CNT + d];
  c[(size_t)n * D_CNT + d] = acc;
}

// -------------------------------- column (node-axis) softmax stats + fusion
__global__ __launch_bounds__(256) void colstats_kernel(
    const float* __restrict__ a1, const float* __restrict__ a2,
    float* __restrict__ cmax, float* __restrict__ csum) {
  __shared__ float red[256];
  const int b = blockIdx.x;                  // [0,128)=a1, [128,256)=a2
  const float* src = (b < D_CNT) ? a1 : a2;
  const int dd = b & (D_CNT - 1);
  const int tid = threadIdx.x;
  float mx = -3.4e38f;
  for (int nn = tid; nn < N_CNT; nn += 256)
    mx = fmaxf(mx, src[(size_t)nn * D_CNT + dd]);
  red[tid] = mx; __syncthreads();
  for (int off = 128; off; off >>= 1) {
    if (tid < off) red[tid] = fmaxf(red[tid], red[tid + off]);
    __syncthreads();
  }
  mx = red[0]; __syncthreads();
  float sm = 0.f;
  for (int nn = tid; nn < N_CNT; nn += 256)
    sm += expf(src[(size_t)nn * D_CNT + dd] - mx);
  red[tid] = sm; __syncthreads();
  for (int off = 128; off; off >>= 1) {
    if (tid < off) red[tid] += red[tid + off];
    __syncthreads();
  }
  if (tid == 0) { cmax[b] = mx; csum[b] = red[0]; }
}

__global__ __launch_bounds__(256) void fusion_kernel(
    const float* __restrict__ a1, const float* __restrict__ a2,
    const float* __restrict__ cmax, const float* __restrict__ csum,
    const float* __restrict__ allemb, const float* __restrict__ atten,
    float* __restrict__ out) {
  const int i = blockIdx.x * 256 + threadIdx.x;
  const int dd = i & (D_CNT - 1);
  const float w1 = expf(a1[i] - cmax[dd]) / csum[dd];
  const float w2 = expf(a2[i] - cmax[D_CNT + dd]) / csum[D_CNT + dd];
  out[i] = w1 * allemb[i] + w2 * atten[i];
}

// ------------------------------------------------------------------ driver
extern "C" void kernel_launch(void* const* d_in, const int* in_sizes, int n_in,
                              void* d_out, int out_size, void* d_ws, size_t ws_size,
                              hipStream_t stream) {
  const float* user = (const float*)d_in[0];
  const float* item = (const float*)d_in[1];
  const int*   rows = (const int*)d_in[2];
  const int*   cols = (const int*)d_in[3];
  const float* vals = (const float*)d_in[4];
  const float* wq   = (const float*)d_in[5];
  const float* bq   = (const float*)d_in[6];
  const float* wk   = (const float*)d_in[7];
  // d_in[8] (bk): constant-in-k score shift, cancels in softmax.
  const float* wv   = (const float*)d_in[9];
  const float* bv   = (const float*)d_in[10];
  const float* fw   = (const float*)d_in[11];
  const float* fb   = (const float*)d_in[12];

  float* out    = (float*)d_out;
  float* outAll = out;                 // [N,D] all_emb
  float* outAtt = out + ND;            // [N,D] atten_embed
  float* outFus = out + 2 * ND;        // [N,D] fusion

  float*  R0    = (float*)d_ws;
  float*  R1    = R0 + ND;
  float*  R2    = R1 + ND;
  __bf16* ego16 = (__bf16*)(R2 + ND);
  int*    topk  = (int*)(ego16 + ND);
  float*  cmax  = (float*)(topk + (size_t)N_CNT * K_TOP);
  float*  csum  = cmax + 2 * D_CNT;

  const dim3 b256(256);
  const int gEW = ND / 256;            // 8192 blocks for elementwise

  // x0 = concat(user, item); sum = 0
  concat_kernel<<<gEW, b256, 0, stream>>>(user, item, R0);
  zero_kernel<<<gEW, b256, 0, stream>>>(R2);
  // layer 1: R0 -> R1
  zero_kernel<<<gEW, b256, 0, stream>>>(R1);
  spmm_kernel<<<E_CNT / 8, b256, 0, stream>>>(rows, cols, vals, R0, R1);
  accum_kernel<<<gEW, b256, 0, stream>>>(R2, R1);
  // layer 2: R1 -> R0
  zero_kernel<<<gEW, b256, 0, stream>>>(R0);
  spmm_kernel<<<E_CNT / 8, b256, 0, stream>>>(rows, cols, vals, R1, R0);
  accum_kernel<<<gEW, b256, 0, stream>>>(R2, R0);
  // layer 3: R0 -> R1  (ego = R1)
  zero_kernel<<<gEW, b256, 0, stream>>>(R1);
  spmm_kernel<<<E_CNT / 8, b256, 0, stream>>>(rows, cols, vals, R0, R1);
  accum_kernel<<<gEW, b256, 0, stream>>>(R2, R1);
  // all_emb = sum/3 ; ego -> bf16 for the ranking GEMM
  mean3_kernel<<<gEW, b256, 0, stream>>>(outAll, R2);
  cvt_bf16_kernel<<<gEW, b256, 0, stream>>>(ego16, R1);
  // fused sim + top-32 (bf16 WMMA, f32 accumulate)
  simtopk_kernel<<<N_CNT / 16, b256, 0, stream>>>(ego16, topk);
  // q = ego@wq^T + bq ; p = q@wk  (so score = p . sample)
  gemm128_kernel<<<N_CNT / 64, b256, 0, stream>>>(R1, wq, bq, nullptr, 0.f, 1, 0, R2);
  gemm128_kernel<<<N_CNT / 64, b256, 0, stream>>>(R2, wk, nullptr, nullptr, 0.f, 0, 0, R0);
  // c[n] = sum_k softmax_k(p.sample) * sample
  attn_kernel<<<N_CNT, dim3(128), 0, stream>>>(R0, R1, topk, R2);
  // atten = c@wv^T + bv + 0.1*ego
  gemm128_kernel<<<N_CNT / 64, b256, 0, stream>>>(R2, wv, bv, R1, 0.1f, 1, 0, outAtt);
  // a1 = tanh(all@fw^T+fb) ; a2 = tanh(atten@fw^T+fb)
  gemm128_kernel<<<N_CNT / 64, b256, 0, stream>>>(outAll, fw, fb, nullptr, 0.f, 1, 1, R0);
  gemm128_kernel<<<N_CNT / 64, b256, 0, stream>>>(outAtt, fw, fb, nullptr, 0.f, 1, 1, R2);
  // node-axis softmax + fusion
  colstats_kernel<<<2 * D_CNT, b256, 0, stream>>>(R0, R2, cmax, csum);
  fusion_kernel<<<gEW, b256, 0, stream>>>(R0, R2, cmax, csum, outAll, outAtt, outFus);
}
